// VectorQuantizer_29403346108847
// MI455X (gfx1250) — matrix operand order repaired and checked by reading, not compiled, not run on hardware
//
#include <hip/hip_runtime.h>
#include <hip/hip_bf16.h>

// ---------------------------------------------------------------------------
// VQ-VAE vector quantizer for gfx1250 (MI455X).
// N=32768 rows (B*H*W), D=256, K=1024 codes.
//
// Phase 1: 32768x1024x256 distance matmul with v_wmma_f32_16x16x32_bf16,
// hi/lo bf16 split of both operands (3 WMMAs per 32-K chunk) for ~fp32
// accurate dot products, fused running argmin.  Codebook chunks are staged
// to LDS by the Tensor Data Mover (tensor_load_to_lds, TENSORcnt), double
// buffered so the DMA overlaps the WMMA stream.  TDM's pad_interval /
// pad_amount insert the 16B-per-row LDS padding (bank-conflict avoidance)
// during the copy itself.
// ---------------------------------------------------------------------------

#define K_CODES 1024
#define DDIM    256
#define HWSZ    1024            // 32*32 spatial
#define BATCH   32
#define NROWS   (BATCH * HWSZ)  // 32768

#define M_TILE  64              // rows per workgroup (4 waves x 16)
#define KC      32              // codes per LDS chunk (double buffered)
#define NCHUNK  (K_CODES / KC)
#define LDA     264             // padded LDS stride in bf16 elems (256 + 8)

typedef __attribute__((ext_vector_type(8)))  __bf16 v8bf;
typedef __attribute__((ext_vector_type(16))) __bf16 v16bf;
typedef __attribute__((ext_vector_type(8)))  float  v8f;
typedef __attribute__((ext_vector_type(4)))  unsigned int u32x4;
typedef __attribute__((ext_vector_type(8)))  int i32x8;
typedef __attribute__((ext_vector_type(4)))  int i32x4;

#if defined(__has_builtin)
#if __has_builtin(__builtin_amdgcn_tensor_load_to_lds) && \
    __has_builtin(__builtin_amdgcn_s_wait_tensorcnt)
#define VQ_USE_TDM 1
#endif
#endif

#if defined(VQ_USE_TDM)
// 2D TDM copy: nrows rows of 512B (128 dword elems) from contiguous global
// memory into LDS at lds_off, inserting 16B (4 dwords) of padding after each
// 128-dword row -> LDS row stride = 528B = LDA bf16 elems.  D# per ISA 8.3/8.4.
__device__ __forceinline__ void tdm_load_2d(unsigned lds_off, const void* gptr,
                                            int nrows) {
    unsigned long long ga = (unsigned long long)(uintptr_t)gptr;
    u32x4 g0;
    g0.x = 0x1u;                                   // count=1, user mode
    g0.y = lds_off;                                // lds_addr (bytes)
    g0.z = (unsigned)ga;                           // global_addr[31:0]
    g0.w = ((unsigned)(ga >> 32) & 0x01FFFFFFu)    // global_addr[56:32]
           | (2u << 30);                           // type=2 ("image")
    i32x8 g1;
    g1[0] = (2 << 16)        // data_size: 4-byte elements
          | (1 << 20)        // pad_enable
          | (6 << 22)        // pad_interval: every 128 dwords
          | (3 << 25);       // pad_amount: 4 dwords (16B)
    g1[1] = (128 << 16);     // tensor_dim0[15:0]=128 at bits 79:48
    g1[2] = (nrows << 16);   // tensor_dim0 hi=0 | tensor_dim1 lo = nrows
    g1[3] = (128 << 16);     // tensor_dim1 hi=0 | tile_dim0 = 128
    g1[4] = nrows & 0xFFFF;  // tile_dim1 = nrows | tile_dim2 = 0 (unused)
    g1[5] = 128;             // tensor_dim0_stride = 128 (contiguous rows)
    g1[6] = 0;
    g1[7] = 0;
    i32x4 z4 = {0, 0, 0, 0};
#if __clang_major__ >= 23
    i32x8 z8 = {0, 0, 0, 0, 0, 0, 0, 0};
    __builtin_amdgcn_tensor_load_to_lds(g0, g1, z4, z4, z8, 0);
#else
    __builtin_amdgcn_tensor_load_to_lds(g0, g1, z4, z4, 0);
#endif
}
#endif  // VQ_USE_TDM

// Assemble a per-lane 16-element bf16 WMMA operand from two 16-byte LDS chunks.
__device__ __forceinline__ v16bf ld16(const __bf16* p0, const __bf16* p1) {
    v8bf a = *(const v8bf*)p0;
    v8bf b = *(const v8bf*)p1;
    v16bf r;
#pragma unroll
    for (int i = 0; i < 8; ++i) { r[i] = a[i]; r[i + 8] = b[i]; }
    return r;
}

// ---------------------------------------------------------------------------
// Prep A: half squared norms of codebook rows: norms[k] = 0.5*||e_k||^2
// ---------------------------------------------------------------------------
__global__ void vq_norms(const float* __restrict__ E, float* __restrict__ norms) {
    int k = blockIdx.x * blockDim.x + threadIdx.x;
    if (k >= K_CODES) return;
    const float* row = E + (size_t)k * DDIM;
    float s = 0.f;
#pragma unroll 8
    for (int d = 0; d < DDIM; ++d) { float v = row[d]; s += v * v; }
    norms[k] = 0.5f * s;
}

// ---------------------------------------------------------------------------
// Prep B: hi/lo bf16 split of the codebook (row-major [K][D], contiguous) so
// phase-1 staging is a pure TDM byte copy.
// ---------------------------------------------------------------------------
__global__ void vq_prep_codebook(const float* __restrict__ E,
                                 __bf16* __restrict__ Ehi,
                                 __bf16* __restrict__ Elo) {
    int i = blockIdx.x * blockDim.x + threadIdx.x;
    if (i >= K_CODES * DDIM) return;
    float e  = E[i];
    __bf16 h = (__bf16)e;
    Ehi[i] = h;
    Elo[i] = (__bf16)(e - (float)h);
}

// ---------------------------------------------------------------------------
// Phase 1: fused distance-matmul + argmin.
// score(n,k) = 0.5*||e_k||^2 - x_n . e_k   (same ranking as the full distance)
// ---------------------------------------------------------------------------
__global__ void __launch_bounds__(128)
vq_phase1(const float*  __restrict__ lat,    // [B, D, H, W] f32
          const __bf16* __restrict__ Ehi,    // [K, D] bf16 (hi)
          const __bf16* __restrict__ Elo,    // [K, D] bf16 (lo)
          const float*  __restrict__ E,      // [K, D] f32 (fallback staging)
          const float*  __restrict__ norms,  // [K] (0.5*||e||^2)
          int*   __restrict__ idx_i,         // [N] workspace
          float* __restrict__ idx_f)         // [N] output (float-cast idx)
{
    __shared__ __bf16 sAhi[M_TILE * LDA];
    __shared__ __bf16 sAlo[M_TILE * LDA];
    __shared__ __bf16 sB[2][2][KC * LDA];    // [buffer][hi/lo][code][d]
    __shared__ float  sNorm[K_CODES];

    const int tid  = threadIdx.x;
    const int lane = tid & 31;
    const int wave = tid >> 5;               // 0..3
    const int lcol    = lane & 15;
    const int halfsel = lane >> 4;           // 0: lanes 0-15, 1: lanes 16-31

    const int n0  = blockIdx.x * M_TILE;
    const int b   = n0 >> 10;                // 64 rows never cross a batch
    const int hw0 = n0 & (HWSZ - 1);

#if defined(VQ_USE_TDM)
    // Kick off chunk 0 DMA before the (longer) A-tile conversion below.
    if (wave == 0) {
        tdm_load_2d((unsigned)(uintptr_t)&sB[0][0][0], Ehi, KC);
        tdm_load_2d((unsigned)(uintptr_t)&sB[0][1][0], Elo, KC);
    }
#endif

    // ---- Stage X tile: hi/lo bf16 split (rows = hw, cols = d) -------------
    for (int i = tid; i < M_TILE * DDIM; i += 128) {
        int r = i & (M_TILE - 1);
        int d = i >> 6;
        float x  = lat[((size_t)b * DDIM + d) * HWSZ + hw0 + r];
        __bf16 h = (__bf16)x;
        sAhi[r * LDA + d] = h;
        sAlo[r * LDA + d] = (__bf16)(x - (float)h);
    }
    // All code half-norms resident in LDS.
    for (int i = tid; i < K_CODES; i += 128) sNorm[i] = norms[i];

#if defined(VQ_USE_TDM)
    if (wave == 0) __builtin_amdgcn_s_wait_tensorcnt(0);
#endif
    __syncthreads();

    // ---- Running per-lane argmin over the 8 C-matrix VGPRs ----------------
    float bestv[8];
    int   besti[8];
#pragma unroll
    for (int r = 0; r < 8; ++r) { bestv[r] = 3.4e38f; besti[r] = 0; }

    const int rowbase = wave * 16;
    const int arow    = rowbase + lcol;

    for (int c = 0; c < NCHUNK; ++c) {
        const int p  = c & 1;
        const int c0 = c * KC;

#if defined(VQ_USE_TDM)
        // Prefetch next chunk into the other buffer while we compute.
        if (wave == 0 && c + 1 < NCHUNK) {
            tdm_load_2d((unsigned)(uintptr_t)&sB[1 - p][0][0],
                        Ehi + (size_t)(c0 + KC) * DDIM, KC);
            tdm_load_2d((unsigned)(uintptr_t)&sB[1 - p][1][0],
                        Elo + (size_t)(c0 + KC) * DDIM, KC);
        }
#else
        // Fallback: synchronous conversion staging into buffer p.
        for (int i = tid; i < KC * DDIM; i += 128) {
            int d    = i & (DDIM - 1);
            int code = i >> 8;
            float e  = E[(size_t)(c0 + code) * DDIM + d];
            __bf16 h = (__bf16)e;
            sB[p][0][code * LDA + d] = h;
            sB[p][1][code * LDA + d] = (__bf16)(e - (float)h);
        }
        __syncthreads();
#endif

        const __bf16* bhiP = &sB[p][0][0];
        const __bf16* bloP = &sB[p][1][0];

        for (int nt = 0; nt < KC / 16; ++nt) {
            v8f acc = {};
#pragma unroll
            for (int kc = 0; kc < DDIM; kc += 32) {
                // A (16x32, MxK): lanes 0-15 hold K {0..7,16..23},
                // lanes 16-31 hold K {8..15,24..31} -> two 16B chunks.
                int aoff = arow * LDA + kc + halfsel * 8;
                v16bf ah = ld16(&sAhi[aoff], &sAhi[aoff + 16]);
                v16bf al = ld16(&sAlo[aoff], &sAlo[aoff + 16]);
                // B (32x16, KxN): lane n = column n; lanes 0-15 K {0..7,16..23},
                // lanes 16-31 K {8..15,24..31} -> two 16B chunks.
                int boff = (nt * 16 + lcol) * LDA + kc + halfsel * 8;
                v16bf bh = ld16(&bhiP[boff], &bhiP[boff + 16]);
                v16bf bl = ld16(&bloP[boff], &bloP[boff + 16]);
                // hi*hi + hi*lo + lo*hi, f32 accumulate: ~fp32 dot product.
                acc = __builtin_amdgcn_wmma_f32_16x16x32_bf16(
                          false, ah, false, bh, (short)0, acc, false, false);
                acc = __builtin_amdgcn_wmma_f32_16x16x32_bf16(
                          false, ah, false, bl, (short)0, acc, false, false);
                acc = __builtin_amdgcn_wmma_f32_16x16x32_bf16(
                          false, al, false, bh, (short)0, acc, false, false);
            }
            // C layout: VGPR r = row (rowbase + r + 8*halfsel), col = lcol.
            float nrm = sNorm[c0 + nt * 16 + lcol];
            int  code = c0 + nt * 16 + lcol;
#pragma unroll
            for (int r = 0; r < 8; ++r) {
                float v = nrm - acc[r];
                if (v < bestv[r]) { bestv[r] = v; besti[r] = code; }
            }
        }

#if defined(VQ_USE_TDM)
        if (wave == 0) __builtin_amdgcn_s_wait_tensorcnt(0);
#endif
        __syncthreads();   // chunk consumed; prefetched buffer visible to all
    }

    // ---- Cross-lane min within each 16-lane half --------------------------
#pragma unroll
    for (int m = 1; m < 16; m <<= 1) {
#pragma unroll
        for (int r = 0; r < 8; ++r) {
            float ov = __shfl_xor(bestv[r], m, 32);
            int   oi = __shfl_xor(besti[r], m, 32);
            if (ov < bestv[r] || (ov == bestv[r] && oi < besti[r])) {
                bestv[r] = ov; besti[r] = oi;
            }
        }
    }

    if (lcol == 0) {
        int row0 = n0 + rowbase + halfsel * 8;
#pragma unroll
        for (int r = 0; r < 8; ++r) {
            idx_i[row0 + r] = besti[r];
            idx_f[row0 + r] = (float)besti[r];
        }
    }
}

// ---------------------------------------------------------------------------
// Phase 2: elementwise loss + quantized output.
//   vq_loss[B,H,W,D] = 1.25*(x-q)^2   (embed + 0.25*commit, same square)
//   q_st   [B,D,H,W] = q              (straight-through forward value)
// ---------------------------------------------------------------------------
__global__ void __launch_bounds__(256)
vq_phase2(const float* __restrict__ lat, const float* __restrict__ E,
          const int* __restrict__ idx,
          float* __restrict__ vq_loss, float* __restrict__ q_st)
{
    __shared__ float tile[32][33];
    const int b   = blockIdx.z;
    const int d0  = blockIdx.y * 32;
    const int hw0 = blockIdx.x * 32;
    const int tx  = threadIdx.x & 31;
    const int ty  = threadIdx.x >> 5;

    const int n    = b * HWSZ + hw0 + tx;
    const int code = idx[n];

    for (int dd = ty; dd < 32; dd += 8) {
        int d    = d0 + dd;
        size_t g = ((size_t)b * DDIM + d) * HWSZ + hw0 + tx;
        float x = lat[g];
        float q = E[(size_t)code * DDIM + d];
        q_st[g] = q;
        float diff = x - q;
        tile[tx][dd] = 1.25f * diff * diff;
    }
    __syncthreads();
    for (int hh = ty; hh < 32; hh += 8) {
        vq_loss[((size_t)(b * HWSZ + hw0 + hh)) * DDIM + d0 + tx] = tile[hh][tx];
    }
}

// ---------------------------------------------------------------------------
extern "C" void kernel_launch(void* const* d_in, const int* in_sizes, int n_in,
                              void* d_out, int out_size, void* d_ws, size_t ws_size,
                              hipStream_t stream) {
    const float* lat = (const float*)d_in[0];   // [32,256,32,32] f32
    const float* E   = (const float*)d_in[1];   // [1024,256]     f32

    float* out     = (float*)d_out;
    float* vq_loss = out;                                   // [B,H,W,D]
    float* q_st    = out + (size_t)NROWS * DDIM;            // [B,D,H,W]
    float* idx_f   = out + 2 * (size_t)NROWS * DDIM;        // [B,H,W] as float

    // workspace layout
    float*  norms = (float*)d_ws;                              // 4 KB
    int*    idx_i = (int*)((char*)d_ws + (4 << 10));           // 128 KB
    __bf16* Ehi   = (__bf16*)((char*)d_ws + (132 << 10));      // 512 KB
    __bf16* Elo   = (__bf16*)((char*)d_ws + (644 << 10));      // 512 KB

    vq_norms<<<4, 256, 0, stream>>>(E, norms);
    vq_prep_codebook<<<(K_CODES * DDIM) / 256, 256, 0, stream>>>(E, Ehi, Elo);
    vq_phase1<<<NROWS / M_TILE, 128, 0, stream>>>(lat, Ehi, Elo, E, norms,
                                                  idx_i, idx_f);
    vq_phase2<<<dim3(HWSZ / 32, DDIM / 32, BATCH), 256, 0, stream>>>(
        lat, E, idx_i, vq_loss, q_st);
}
